// KGTOREModel_64604898066610
// MI455X (gfx1250) — matrix operand order repaired and checked by reading, not compiled, not run on hardware
//
#include <hip/hip_runtime.h>
#include <hip/hip_bf16.h>

typedef __attribute__((ext_vector_type(16))) __bf16 v16bf;
typedef __attribute__((ext_vector_type(8)))  float  v8f;

#define U_DIM 100000
#define I_DIM 50000
#define N_DIM 150000   // U + I
#define E_DIM 1000000
#define K_DIM 64
#define B_DIM 16384

// ---------------------------------------------------------------------------
// fw[j] = sum_k F[j,k] * proj_w[k]   (fold F @ proj_w into one 64-vector)
// fw[64] = proj_b
// ---------------------------------------------------------------------------
__global__ void k_fold_fw(const float* __restrict__ F,
                          const float* __restrict__ pw,
                          const float* __restrict__ pb,
                          float* __restrict__ fw) {
    int j = threadIdx.x;   // 64 threads
    float s = 0.f;
    #pragma unroll 8
    for (int k = 0; k < 64; ++k) s += F[j * 64 + k] * pw[k];
    fw[j] = s;
    if (j == 0) fw[64] = pb[0];
}

// ---------------------------------------------------------------------------
// ew[e] = dot(edge_features[e,:], fw) + bias, 16 edges per wave via
// v_wmma_f32_16x16x32_bf16. B matrix: every column = fw chunk, so D[:,0]
// holds the 16 edge weights (lanes 0 and 16 of the 8 accumulator VGPRs).
// ---------------------------------------------------------------------------
__global__ void k_ew_wmma(const float* __restrict__ ef,
                          const float* __restrict__ fw,
                          float* __restrict__ ew,
                          int ntiles) {
    int wave = (int)((blockIdx.x * blockDim.x + threadIdx.x) >> 5);
    if (wave >= ntiles) return;                 // wave-uniform branch
    int lane  = threadIdx.x & 31;
    int r0    = wave << 4;                      // 16 rows per tile
    int row   = r0 + (lane & 15);
    int kb    = (lane >> 4) << 3;               // 0 for lanes 0-15, 8 for 16-31

    const float* rp = ef + (long)row * K_DIM;

    v16bf a0, a1, b0, b1;
    #pragma unroll
    for (int i = 0; i < 8; ++i) {
        // A 16x32 bf16 layout: elems 0..7 -> K=kb..kb+7, elems 8..15 -> K=kb+16..kb+23
        a0[i]     = (__bf16)rp[kb + i];
        a0[i + 8] = (__bf16)rp[kb + 16 + i];
        a1[i]     = (__bf16)rp[32 + kb + i];
        a1[i + 8] = (__bf16)rp[32 + kb + 16 + i];
    }
    #pragma unroll
    for (int i = 0; i < 8; ++i) {
        // B 32x16 bf16: same K order as A; all columns identical = fw
        b0[i]     = (__bf16)fw[kb + i];
        b0[i + 8] = (__bf16)fw[kb + 16 + i];
        b1[i]     = (__bf16)fw[32 + kb + i];
        b1[i + 8] = (__bf16)fw[32 + kb + 16 + i];
    }

    v8f c = {};
    c = __builtin_amdgcn_wmma_f32_16x16x32_bf16(false, a0, false, b0,
                                                (short)0, c, false, false);
    c = __builtin_amdgcn_wmma_f32_16x16x32_bf16(false, a1, false, b1,
                                                (short)0, c, false, false);

    float bias = fw[64];
    // D layout: VGPR r, lane 0 -> (M=r, N=0); lane 16 -> (M=r+8, N=0)
    if (lane == 0) {
        #pragma unroll
        for (int r = 0; r < 8; ++r) ew[r0 + r] = c[r] + bias;
    } else if (lane == 16) {
        #pragma unroll
        for (int r = 0; r < 8; ++r) ew[r0 + 8 + r] = c[r] + bias;
    }
}

// ---------------------------------------------------------------------------
// deg[dst[e]] += 1   (hardware f32 atomic in L2)
// ---------------------------------------------------------------------------
__global__ void k_degree(const int* __restrict__ dst, float* __restrict__ deg,
                         int twoE) {
    int e = blockIdx.x * blockDim.x + threadIdx.x;
    if (e < twoE) unsafeAtomicAdd(&deg[dst[e]], 1.0f);
}

// in place: deg -> dinv
__global__ void k_dinv(float* __restrict__ deg, int n) {
    int i = blockIdx.x * blockDim.x + threadIdx.x;
    if (i < n) {
        float d = deg[i];
        deg[i] = (d > 0.f) ? rsqrtf(fmaxf(d, 1.0f)) : 0.f;
    }
}

__global__ void k_coef(const int* __restrict__ src, const int* __restrict__ dst,
                       const float* __restrict__ dinv, const float* __restrict__ ew,
                       float* __restrict__ coef, int twoE) {
    int e = blockIdx.x * blockDim.x + threadIdx.x;
    if (e < twoE) {
        int ee = (e < E_DIM) ? e : e - E_DIM;   // ew duplicated per direction
        coef[e] = dinv[src[e]] * dinv[dst[e]] * ew[ee];
    }
}

// x0 = concat(Gu, Gi); acc = ALPHA[0] * x0 (= x0)
__global__ void k_init_x(const float* __restrict__ Gu, const float* __restrict__ Gi,
                         float* __restrict__ x, float* __restrict__ acc, int total) {
    int i = blockIdx.x * blockDim.x + threadIdx.x;
    if (i < total) {
        float v = (i < U_DIM * K_DIM) ? Gu[i] : Gi[i - U_DIM * K_DIM];
        x[i] = v;
        acc[i] = v;
    }
}

// xout[dst[e], k] += coef[e] * xin[src[e], k]; thread = (edge, channel)
__global__ void k_spmm(const int* __restrict__ src, const int* __restrict__ dst,
                       const float* __restrict__ coef,
                       const float* __restrict__ xin, float* __restrict__ xout,
                       unsigned twoE) {
    unsigned t = blockIdx.x * blockDim.x + threadIdx.x;
    unsigned e = t >> 6;
    if (e >= twoE) return;
    int k = t & 63;
    float v = coef[e] * xin[(long)src[e] * K_DIM + k];
    unsafeAtomicAdd(&xout[(long)dst[e] * K_DIM + k], v);
}

__global__ void k_axpy(float* __restrict__ acc, const float* __restrict__ x,
                       float alpha, int total) {
    int i = blockIdx.x * blockDim.x + threadIdx.x;
    if (i < total) acc[i] += alpha * x[i];
}

// out[b] = dot(acc[users[b]], acc[U + items[b]]); one wave32 per pair
__global__ void k_score(const float* __restrict__ acc,
                        const int* __restrict__ users, const int* __restrict__ items,
                        float* __restrict__ out, int B) {
    int t = blockIdx.x * blockDim.x + threadIdx.x;
    int b = t >> 5;
    if (b >= B) return;
    int lane = t & 31;
    const float* gu = acc + (long)users[b] * K_DIM;
    const float* gi = acc + ((long)U_DIM + items[b]) * K_DIM;
    float s = gu[lane] * gi[lane] + gu[lane + 32] * gi[lane + 32];
    #pragma unroll
    for (int off = 16; off > 0; off >>= 1) s += __shfl_xor(s, off, 32);
    if (lane == 0) out[b] = s;
}

// ---------------------------------------------------------------------------
extern "C" void kernel_launch(void* const* d_in, const int* in_sizes, int n_in,
                              void* d_out, int out_size, void* d_ws, size_t ws_size,
                              hipStream_t stream) {
    const float* Gu = (const float*)d_in[0];           // [U, 64]
    const float* Gi = (const float*)d_in[1];           // [I, 64]
    const float* F  = (const float*)d_in[2];           // [64, 64]
    const float* pw = (const float*)d_in[3];           // [64, 1]
    const float* pb = (const float*)d_in[4];           // [1]
    const float* ef = (const float*)d_in[5];           // [E, 64]
    const int*   ei = (const int*)d_in[6];             // [2, 2E]
    const int*   users = (const int*)d_in[7];          // [B]
    const int*   items = (const int*)d_in[8];          // [B]
    float* out = (float*)d_out;                        // [B]

    const int twoE = 2 * E_DIM;
    const int totalX = N_DIM * K_DIM;
    const int* src = ei;                               // row 0
    const int* dst = ei + twoE;                        // row 1

    // workspace layout (floats)
    float* ws   = (float*)d_ws;
    float* fw   = ws;                                  // 65 (padded to 256)
    float* ew   = ws + 256;                            // E
    float* dinv = ew + E_DIM;                          // N (deg -> dinv in place)
    float* coef = dinv + N_DIM;                        // 2E
    float* x0   = coef + twoE;                         // N*64
    float* x1   = x0 + totalX;                         // N*64
    float* acc  = x1 + totalX;                         // N*64

    // 1) fold F @ proj_w
    k_fold_fw<<<1, 64, 0, stream>>>(F, pw, pb, fw);

    // 2) edge weights via WMMA (E/16 tiles, 8 waves per 256-thread block)
    {
        int ntiles = E_DIM / 16;                       // 62500
        int blocks = (ntiles + 7) / 8;
        k_ew_wmma<<<blocks, 256, 0, stream>>>(ef, fw, ew, ntiles);
    }

    // 3) degrees -> dinv
    hipMemsetAsync(dinv, 0, (size_t)N_DIM * sizeof(float), stream);
    k_degree<<<(twoE + 255) / 256, 256, 0, stream>>>(dst, dinv, twoE);
    k_dinv<<<(N_DIM + 255) / 256, 256, 0, stream>>>(dinv, N_DIM);

    // 4) per-edge coefficients
    k_coef<<<(twoE + 255) / 256, 256, 0, stream>>>(src, dst, dinv, ew, coef, twoE);

    // 5) x0 = concat(Gu, Gi); acc = x0
    k_init_x<<<(totalX + 255) / 256, 256, 0, stream>>>(Gu, Gi, x0, acc, totalX);

    // 6) 3 propagation layers (L2-resident gather/scatter)
    const float alphas[4] = {1.0f, 0.5f, 1.0f / 3.0f, 0.25f};
    float* xin = x0;
    float* xout = x1;
    unsigned spmmThreads = (unsigned)twoE * 64u;
    unsigned spmmBlocks = (spmmThreads + 255u) / 256u;
    for (int l = 0; l < 3; ++l) {
        hipMemsetAsync(xout, 0, (size_t)totalX * sizeof(float), stream);
        k_spmm<<<spmmBlocks, 256, 0, stream>>>(src, dst, coef, xin, xout,
                                               (unsigned)twoE);
        k_axpy<<<(totalX + 255) / 256, 256, 0, stream>>>(acc, xout,
                                                         alphas[l + 1], totalX);
        float* tmp = xin; xin = xout; xout = tmp;
    }

    // 7) masked dot-product scoring, one wave per (user, item)
    {
        int threads = B_DIM * 32;
        k_score<<<(threads + 255) / 256, 256, 0, stream>>>(acc, users, items,
                                                           out, B_DIM);
    }
}
